// Convolution_21191368639117
// MI455X (gfx1250) — matrix operand order repaired and checked by reading, not compiled, not run on hardware
//
#include <hip/hip_runtime.h>
#include <hip/hip_bf16.h>

// ---- problem constants (match reference) ----
static constexpr int NPTS = 512;
static constexpr int CIO  = 16;   // C_IN == C_OUT == 16
static constexpr int NRAD = 8;
static constexpr int NANG = 4;
static constexpr int NRM  = 32;   // NRAD * NANG
static constexpr int BT   = 32;   // b-chunk == WMMA K

typedef __bf16        v16bf __attribute__((ext_vector_type(16)));
typedef __bf16        v2bf  __attribute__((ext_vector_type(2)));
typedef float         v8f   __attribute__((ext_vector_type(8)));
typedef float         v4f   __attribute__((ext_vector_type(4)));
typedef unsigned int  v4u   __attribute__((ext_vector_type(4)));

union AFrag { v16bf v; unsigned int u[8]; };

// LDS: basis tiles (phase 1) and G spill (phase 2) alias each other.
// Stride 16 (no pad) so each lane's 4-DWORD fragment groups are 16B-aligned
// and contiguous -> ds_load_b128.
union alignas(16) ShU {
  unsigned int P[NRM][16][16];  // packed bf16 pairs along b: [rm][a][b/2]
  float        G[NRM][16][16];  // f32 G[rm][a][j] after the b loop
};

// f32 pair -> packed bf16 DWORD; backend lowers to v_cvt_pk_bf16_f32 when
// available (else a short cvt sequence) instead of 6-op manual rounding.
__device__ __forceinline__ unsigned int pack_bf16(float lo, float hi) {
  v2bf p;
  p[0] = (__bf16)lo;
  p[1] = (__bf16)hi;
  unsigned int r;
  __builtin_memcpy(&r, &p, sizeof(r));
  return r;
}

__device__ __forceinline__ v8f v8f_zero() {
  v8f r;
  #pragma unroll
  for (int i = 0; i < 8; ++i) r[i] = 0.0f;
  return r;
}

extern "C" __global__ __launch_bounds__(256)
void econv_wmma(const float* __restrict__ feat,
                const float* __restrict__ geo,
                const float* __restrict__ W,
                float* __restrict__ out) {
  __shared__ ShU sh;
  __shared__ alignas(16) unsigned int shF[16][16];  // [j][K/2] packed bf16 pairs

  const int t    = threadIdx.x;
  const int lane = t & 31;
  const int wave = t >> 5;              // 8 waves; wave owns rm = 4*wave..4*wave+3
  const int z    = blockIdx.x >> 5;     // 32 a-tiles per batch
  const int a0   = (blockIdx.x & 31) << 4;

  // basis work assignment: thread -> (a_local, b-pair)
  const int al = t & 15;
  const int bp = t >> 4;                // b_local = 2*bp, 2*bp+1

  const float* gz = geo + (size_t)z * NPTS * 3;
  const float* fz = feat + (size_t)z * NPTS * CIO;

  // warm L2/WGP$ for the small resident operands (global_prefetch_b8)
  __builtin_prefetch(W + (size_t)t * 32, 0, 0);       // 32 KB of weights
  __builtin_prefetch(fz + (size_t)t * 32, 0, 0);      // 32 KB of features (this z)

  const float gax = gz[(a0 + al) * 3 + 0];
  const float gay = gz[(a0 + al) * 3 + 1];
  const float gaz = gz[(a0 + al) * 3 + 2];

  // fragment lane coordinates
  const int ca = lane & 15;
  const int ch = lane >> 4;

  v8f acc[4];
  #pragma unroll
  for (int q = 0; q < 4; ++q) acc[q] = v8f_zero();

  for (int b0 = 0; b0 < NPTS; b0 += BT) {
    __syncthreads();  // previous chunk's fragment loads done before overwrite

    // ---- basis for pairs (a0+al, b0+2bp) and (a0+al, b0+2bp+1) ----
    const int b = b0 + 2 * bp;
    const float rx0 = gz[b * 3 + 0] - gax, ry0 = gz[b * 3 + 1] - gay, rz0 = gz[b * 3 + 2] - gaz;
    const float rx1 = gz[b * 3 + 3] - gax, ry1 = gz[b * 3 + 4] - gay, rz1 = gz[b * 3 + 5] - gaz;
    const float d20 = rx0 * rx0 + ry0 * ry0 + rz0 * rz0 + 1e-12f;
    const float d21 = rx1 * rx1 + ry1 * ry1 + rz1 * rz1 + 1e-12f;
    const float inv0 = __builtin_amdgcn_rsqf(d20);
    const float inv1 = __builtin_amdgcn_rsqf(d21);
    const float d0 = d20 * inv0, d1 = d21 * inv1;     // = sqrt(d2)
    const float ang0[4] = {1.0f, rx0 * inv0, ry0 * inv0, rz0 * inv0};
    const float ang1[4] = {1.0f, rx1 * inv1, ry1 * inv1, rz1 * inv1};
    #pragma unroll
    for (int r = 0; r < NRAD; ++r) {
      const float c = (3.0f / 7.0f) * (float)r;       // linspace(0,3,8)
      const float u0 = d0 - c, u1 = d1 - c;
      const float rad0 = __expf(-8.0f * u0 * u0);
      const float rad1 = __expf(-8.0f * u1 * u1);
      #pragma unroll
      for (int m = 0; m < NANG; ++m)
        sh.P[r * NANG + m][al][bp] = pack_bf16(rad0 * ang0[m], rad1 * ang1[m]);
    }

    // ---- stage feature tile [32b x 16j] as packed bf16 K-pairs, [j][K/2] ----
    {
      const float f0 = fz[(b0 + 2 * bp) * CIO + al];
      const float f1 = fz[(b0 + 2 * bp + 1) * CIO + al];
      shF[al][bp] = pack_bf16(f0, f1);
    }
    __syncthreads();

    // B fragment (shared by this wave's 4 WMMAs): K = (v>=4?16:0) + ch*8 + (v&3)*2 + p, N = ca
    AFrag bfr;
    {
      const v4u blo = *(const v4u*)&shF[ca][ch * 4];      // ds_load_b128
      const v4u bhi = *(const v4u*)&shF[ca][8 + ch * 4];  // ds_load_b128
      #pragma unroll
      for (int v = 0; v < 4; ++v) { bfr.u[v] = blo[v]; bfr.u[v + 4] = bhi[v]; }
    }

    // A fragments + WMMA per owned rm: K = (v>=4?16:0) + ch*8 + (v&3)*2 + p
    #pragma unroll
    for (int q = 0; q < 4; ++q) {
      const int rm = wave * 4 + q;
      AFrag afr;
      const v4u alo = *(const v4u*)&sh.P[rm][ca][ch * 4];      // kp = ch*4 + 0..3
      const v4u ahi = *(const v4u*)&sh.P[rm][ca][8 + ch * 4];  // kp = 8 + ch*4 + 0..3
      #pragma unroll
      for (int v = 0; v < 4; ++v) { afr.u[v] = alo[v]; afr.u[v + 4] = ahi[v]; }
      acc[q] = __builtin_amdgcn_wmma_f32_16x16x32_bf16(
          false, afr.v, false, bfr.v, (short)0, acc[q], false, false);
    }
  }

  __syncthreads();  // all waves done reading P before we alias it as G

  // spill G[rm][a][j] from C-layout accumulators (M = v + 8*ch, N = ca)
  #pragma unroll
  for (int q = 0; q < 4; ++q) {
    const int rm = wave * 4 + q;
    #pragma unroll
    for (int v = 0; v < 8; ++v)
      sh.G[rm][v + ch * 8][ca] = acc[q][v];
  }
  __syncthreads();

  // ---- second GEMM: out[a,i] = sum_{rm,j} G[a,rm,j] * W[rm,i,j] (K = 512) ----
  if (wave == 0) {
    v8f o = v8f_zero();
    #pragma unroll 4
    for (int pr = 0; pr < 16; ++pr) {         // two rm per K=32 step
      const int rm0 = pr * 2;
      AFrag afr, bfr;
      // A: M = a = ca; rm = rm0 + ch, j = 2v + p
      const v4f g0a = *(const v4f*)&sh.G[rm0 + ch][ca][0];
      const v4f g0b = *(const v4f*)&sh.G[rm0 + ch][ca][4];
      const v4f g1a = *(const v4f*)&sh.G[rm0 + ch][ca][8];
      const v4f g1b = *(const v4f*)&sh.G[rm0 + ch][ca][12];
      afr.u[0] = pack_bf16(g0a[0], g0a[1]);
      afr.u[1] = pack_bf16(g0a[2], g0a[3]);
      afr.u[2] = pack_bf16(g0b[0], g0b[1]);
      afr.u[3] = pack_bf16(g0b[2], g0b[3]);
      afr.u[4] = pack_bf16(g1a[0], g1a[1]);
      afr.u[5] = pack_bf16(g1a[2], g1a[3]);
      afr.u[6] = pack_bf16(g1b[0], g1b[1]);
      afr.u[7] = pack_bf16(g1b[2], g1b[3]);
      // B: N = i = ca; K = ch*16 + 2v + p -> row W[rm0+ch][ca][0..15]
      const float* wrow = W + ((size_t)(rm0 + ch) * 16 + ca) * 16;
      const v4f w0 = *(const v4f*)(wrow + 0);
      const v4f w1 = *(const v4f*)(wrow + 4);
      const v4f w2 = *(const v4f*)(wrow + 8);
      const v4f w3 = *(const v4f*)(wrow + 12);
      bfr.u[0] = pack_bf16(w0[0], w0[1]);
      bfr.u[1] = pack_bf16(w0[2], w0[3]);
      bfr.u[2] = pack_bf16(w1[0], w1[1]);
      bfr.u[3] = pack_bf16(w1[2], w1[3]);
      bfr.u[4] = pack_bf16(w2[0], w2[1]);
      bfr.u[5] = pack_bf16(w2[2], w2[3]);
      bfr.u[6] = pack_bf16(w3[0], w3[1]);
      bfr.u[7] = pack_bf16(w3[2], w3[3]);
      o = __builtin_amdgcn_wmma_f32_16x16x32_bf16(
          false, afr.v, false, bfr.v, (short)0, o, false, false);
    }
    const float scale = 0.044194173824159216f;  // 1/sqrt(512)
    #pragma unroll
    for (int v = 0; v < 8; ++v)
      out[((size_t)z * NPTS + a0 + v + ch * 8) * CIO + ca] = o[v] * scale;
  }
}

extern "C" void kernel_launch(void* const* d_in, const int* in_sizes, int n_in,
                              void* d_out, int out_size, void* d_ws, size_t ws_size,
                              hipStream_t stream) {
  (void)in_sizes; (void)n_in; (void)out_size; (void)d_ws; (void)ws_size;
  const float* feat = (const float*)d_in[0];   // [4,512,16]
  const float* geo  = (const float*)d_in[1];   // [4,512,3]
  const float* W    = (const float*)d_in[2];   // [8,4,16,16]
  float* out        = (float*)d_out;           // [4,512,16]
  dim3 grid(4 * (NPTS / 16));                  // 128 blocks: (z, a-tile)
  dim3 block(256);                             // 8 wave32
  hipLaunchKernelGGL(econv_wmma, grid, block, 0, stream, feat, geo, W, out);
}
